// SlidingWindowCausalAttention_11003706212959
// MI455X (gfx1250) — matrix operand order repaired and checked by reading, not compiled, not run on hardware
//
#include <hip/hip_runtime.h>

// ---------------------------------------------------------------------------
// Sliding-window causal attention, MI455X (gfx1250), wave32 flash-attention.
// B=1 H=16 T=8192 D=64 W=1024. fp32 in/out, f16 WMMA compute (f32 accum).
//
// Roofline: 69 GFLOP vs ~134MB HBM -> memory bound (~6us floor at 23.3TB/s).
// K/V re-reads across q-tiles are L2 traffic (K16+Vt16 = 32MB << 192MB L2).
//  * prep kernel: K -> f16 row-major, V -> f16 transposed [h][d][t] in d_ws.
//  * attention: 4 waves share one KV loop; wave0 DMAs K/Vt f16 tiles into
//    LDS via TENSOR_LOAD_TO_LDS (double buffered, TDM row padding for
//    conflict-free ds_load_b128), synced by s_wait_tensorcnt + barriers.
//  * softmax in log2 domain: SCALE*log2e folded into Q, native v_exp_f32;
//    row-sum of P computed by an extra WMMA against an all-ones matrix
//    (result lands replicated in C/D layout), killing 32 bpermutes/iter.
// ---------------------------------------------------------------------------

typedef _Float16 half_t;
typedef __attribute__((ext_vector_type(16))) _Float16 v16h;
typedef __attribute__((ext_vector_type(8)))  _Float16 v8h;
typedef __attribute__((ext_vector_type(8)))  float    v8f;
typedef unsigned int u32x4 __attribute__((ext_vector_type(4)));
typedef int          i32x8 __attribute__((ext_vector_type(8)));
typedef int          i32x4 __attribute__((ext_vector_type(4)));

#define NH      16
#define T_SEQ   8192
#define DH      64
#define WIN     1024
#define WAVES   4
#define MW      16            // query rows per wave
#define QB      (MW*WAVES)    // 64 query rows per block
#define KV_STEP 32
#define KROW_H  72            // halves per K LDS row   (64 data + 8 pad = 144B)
#define VROW_H  36            // halves per Vt LDS row  (32 data + 4 pad = 72B)
#define PSTRIDE 40            // halves per P LDS row (padded)
#define NEGI    (-1.0e30f)
#define QSCALE  0.1803368801111244f   // (1/sqrt(64)) * log2(e): scores in log2

// ---------------------------------------------------------------------------
// prep: K -> f16 row-major [h][t][d]; V -> f16 transposed [h][d][t]
// ---------------------------------------------------------------------------
__global__ __launch_bounds__(256)
void swa_prep_kernel(const float* __restrict__ K, const float* __restrict__ V,
                     half_t* __restrict__ K16, half_t* __restrict__ Vt16)
{
  __shared__ half_t vt[64][65];               // padded transpose tile
  const int nTT = T_SEQ / 64;                 // 128 t-tiles
  const int h  = blockIdx.x / nTT;
  const int t0 = (blockIdx.x % nTT) * 64;

  const float* __restrict__ Kh = K + (size_t)h * T_SEQ * DH;
  const float* __restrict__ Vh = V + (size_t)h * T_SEQ * DH;
  half_t* __restrict__ K16h  = K16  + (size_t)h * T_SEQ * DH;
  half_t* __restrict__ Vt16h = Vt16 + (size_t)h * T_SEQ * DH;

#pragma unroll
  for (int i = 0; i < 16; ++i) {
    const int idx = i * 256 + threadIdx.x;    // 0..4095
    const int t = idx >> 6, d = idx & 63;     // coalesced over d
    K16h[(size_t)(t0 + t) * DH + d] = (half_t)Kh[(size_t)(t0 + t) * DH + d];
    vt[t][d] = (half_t)Vh[(size_t)(t0 + t) * DH + d];
  }
  __syncthreads();
#pragma unroll
  for (int i = 0; i < 16; ++i) {
    const int idx = i * 256 + threadIdx.x;
    const int d = idx >> 6, t = idx & 63;     // coalesced over t
    Vt16h[(size_t)d * T_SEQ + t0 + t] = vt[t][d];
  }
}

// ---------------------------------------------------------------------------
// TDM helper: 2D f16 tile -> LDS (clang-23 6-arg builtin; D# per ISA 08 §8)
// ---------------------------------------------------------------------------
static __device__ inline void tdm_load_2d(unsigned lds_off, const half_t* gptr,
                                          unsigned g1dw0, unsigned d1, unsigned d2,
                                          unsigned d3, unsigned d4, unsigned d5)
{
  const unsigned long long ga = (unsigned long long)(uintptr_t)gptr;
  u32x4 g0;
  g0[0] = 1u;                                          // count=1, user mode
  g0[1] = lds_off;                                     // lds_addr (bytes)
  g0[2] = (unsigned)(ga & 0xFFFFFFFFu);                // global_addr[31:0]
  g0[3] = (unsigned)((ga >> 32) & 0x01FFFFFFu) | (2u << 30);  // addr[56:32]|type=2
  i32x8 g1;
  g1[0] = (int)g1dw0;   // mask=0 | data_size=1(2B)<<16 | pad_enable<<20 | pad cfg
  g1[1] = (int)d1;      // abar=0 | tensor_dim0.lo16 << 16
  g1[2] = (int)d2;      // tensor_dim0.hi16 | tensor_dim1.lo16 << 16
  g1[3] = (int)d3;      // tensor_dim1.hi16 | tile_dim0 << 16
  g1[4] = (int)d4;      // tile_dim1 | tile_dim2(0) << 16
  g1[5] = (int)d5;      // tensor_dim0_stride[31:0]
  g1[6] = 0;            // stride0.hi16 | tensor_dim1_stride.lo16
  g1[7] = 0;
  const i32x4 z4 = {0, 0, 0, 0};                       // groups 2/3: 2D tensor
  const i32x8 z8 = {0, 0, 0, 0, 0, 0, 0, 0};
  __builtin_amdgcn_tensor_load_to_lds(g0, g1, z4, z4, z8, 0);
}

// group1 dword0 configs: data_size=2B, pad_enable, pad_interval/amount codes
// K : rows of 64 halves = 32 dw, pad 4 dw  -> interval code 4, amount code 3
// Vt: rows of 32 halves = 16 dw, pad 2 dw  -> interval code 3, amount code 1
#define G1DW0_K  ((1u<<16) | (1u<<20) | (4u<<22) | (3u<<25))
#define G1DW0_V  ((1u<<16) | (1u<<20) | (3u<<22) | (1u<<25))

static __device__ inline v16h ld32h(const half_t* p) {
  v8h a = *reinterpret_cast<const v8h*>(p);
  v8h b = *reinterpret_cast<const v8h*>(p + 16);
  v16h r;
#pragma unroll
  for (int i = 0; i < 8; ++i) { r[i] = a[i]; r[8 + i] = b[i]; }
  return r;
}

// 8 consecutive f32, scaled, -> halves [o..o+7] of a v16h
static __device__ inline void cvt8s(const float* __restrict__ p, v16h& d,
                                    const int o, const float s) {
  float4 a = reinterpret_cast<const float4*>(p)[0];
  float4 b = reinterpret_cast<const float4*>(p)[1];
  d[o+0]=(half_t)(a.x*s); d[o+1]=(half_t)(a.y*s); d[o+2]=(half_t)(a.z*s); d[o+3]=(half_t)(a.w*s);
  d[o+4]=(half_t)(b.x*s); d[o+5]=(half_t)(b.y*s); d[o+6]=(half_t)(b.z*s); d[o+7]=(half_t)(b.w*s);
}

static __device__ inline float rmax16(float v) {
#pragma unroll
  for (int m = 8; m >= 1; m >>= 1) v = fmaxf(v, __shfl_xor(v, m, 32));
  return v;
}

// ---------------------------------------------------------------------------
// main attention kernel
// ---------------------------------------------------------------------------
__global__ __launch_bounds__(WAVES * 32)
void swa_fwd_kernel(const float* __restrict__ Q,
                    const half_t* __restrict__ K16,
                    const half_t* __restrict__ Vt16,
                    float* __restrict__ O)
{
  __shared__ half_t lds_k[2][KV_STEP * KROW_H];   // 2 x 4608B, TDM-padded rows
  __shared__ half_t lds_v[2][DH * VROW_H];        // 2 x 4608B, TDM-padded rows
  __shared__ half_t lds_p[WAVES][16 * PSTRIDE];   // per-wave P transpose

  const int lane = threadIdx.x & 31;
  const int wave = threadIdx.x >> 5;
  const int n    = lane & 15;
  const int hi   = lane >> 4;
  const int moff = hi * 8;

  const int nQT = T_SEQ / QB;                     // 128 q-tiles per head
  const int h   = blockIdx.x / nQT;
  const int qb  = (blockIdx.x % nQT) * QB;        // block's first query row
  const int q0  = qb + wave * MW;                 // wave's first query row

  const float*  __restrict__ Qh  = Q    + (size_t)h * T_SEQ * DH;
  const half_t* __restrict__ Kh  = K16  + (size_t)h * T_SEQ * DH;
  const half_t* __restrict__ Vth = Vt16 + (size_t)h * T_SEQ * DH;
  float*        __restrict__ Oh  = O    + (size_t)h * T_SEQ * DH;

  // ---- Q tile (16x64), pre-scaled by (1/sqrt(D))*log2(e), into A layout --
  v16h qa0, qa1;
  {
    const float* qrow = Qh + (size_t)(q0 + n) * DH;
    cvt8s(qrow +  0 + hi * 8, qa0, 0, QSCALE);
    cvt8s(qrow + 16 + hi * 8, qa0, 8, QSCALE);
    cvt8s(qrow + 32 + hi * 8, qa1, 0, QSCALE);
    cvt8s(qrow + 48 + hi * 8, qa1, 8, QSCALE);
  }

  // all-ones B matrix: l_acc = P @ ones lands replicated in C/D layout
  v16h ones;
#pragma unroll
  for (int i = 0; i < 16; ++i) ones[i] = (half_t)1.0f;

  v8f   acc[4];
  v8f   l_acc = (v8f){0,0,0,0,0,0,0,0};
  float m_i[8];
#pragma unroll
  for (int dt = 0; dt < 4; ++dt) acc[dt] = (v8f){0,0,0,0,0,0,0,0};
#pragma unroll
  for (int r = 0; r < 8; ++r) m_i[r] = NEGI;

  const int jb = (qb >= WIN) ? (qb - WIN) : 0;    // 64-aligned
  const int je = qb + QB;

  // descriptor constants
  const unsigned kD1 = ((unsigned)DH) << 16;                       // dim0=64
  const unsigned kD2 = ((unsigned)T_SEQ & 0xFFFFu) << 16;          // dim1=8192
  const unsigned kD3 = ((unsigned)DH) << 16;                       // tile0=64
  const unsigned kD4 = (unsigned)KV_STEP;                          // tile1=32
  const unsigned kD5 = (unsigned)DH;                               // stride0=64
  const unsigned vD1 = ((unsigned)T_SEQ & 0xFFFFu) << 16;          // dim0=8192
  const unsigned vD2 = ((unsigned)DH) << 16;                       // dim1=64
  const unsigned vD3 = ((unsigned)KV_STEP) << 16;                  // tile0=32
  const unsigned vD4 = (unsigned)DH;                               // tile1=64
  const unsigned vD5 = (unsigned)T_SEQ;                            // stride0=8192

  // prologue: DMA first KV tile into buffer 0
  if (wave == 0) {
    tdm_load_2d((unsigned)(uintptr_t)&lds_k[0][0], Kh + (size_t)jb * DH,
                G1DW0_K, kD1, kD2, kD3, kD4, kD5);
    tdm_load_2d((unsigned)(uintptr_t)&lds_v[0][0], Vth + jb,
                G1DW0_V, vD1, vD2, vD3, vD4, vD5);
  }

  int cur = 0;
  for (int j = jb; j < je; j += KV_STEP, cur ^= 1) {
    if (wave == 0) __builtin_amdgcn_s_wait_tensorcnt(0);
    __syncthreads();                       // buf[cur] ready for all waves

    if (wave == 0 && j + KV_STEP < je) {   // overlap next DMA with compute
      tdm_load_2d((unsigned)(uintptr_t)&lds_k[cur ^ 1][0],
                  Kh + (size_t)(j + KV_STEP) * DH, G1DW0_K, kD1, kD2, kD3, kD4, kD5);
      tdm_load_2d((unsigned)(uintptr_t)&lds_v[cur ^ 1][0],
                  Vth + (j + KV_STEP), G1DW0_V, vD1, vD2, vD3, vD4, vD5);
    }

    // skip tiles fully outside this wave's window (barriers stay uniform)
    if (j <= q0 + MW - 1 && j + KV_STEP - 1 >= q0 - WIN) {
      // ---- S = Q @ K^T (two 16x16 tiles) from LDS; S already log2-scaled -
      const half_t* kb = &lds_k[cur][0];
      v8f s0 = {0,0,0,0,0,0,0,0}, s1 = {0,0,0,0,0,0,0,0};
      {
        const half_t* r0 = kb + (size_t)n        * KROW_H + 8 * hi;
        const half_t* r1 = kb + (size_t)(n + 16) * KROW_H + 8 * hi;
        s0 = __builtin_amdgcn_wmma_f32_16x16x32_f16(false, qa0, false, ld32h(r0),      (short)0, s0, false, false);
        s0 = __builtin_amdgcn_wmma_f32_16x16x32_f16(false, qa1, false, ld32h(r0 + 32), (short)0, s0, false, false);
        s1 = __builtin_amdgcn_wmma_f32_16x16x32_f16(false, qa0, false, ld32h(r1),      (short)0, s1, false, false);
        s1 = __builtin_amdgcn_wmma_f32_16x16x32_f16(false, qa1, false, ld32h(r1 + 32), (short)0, s1, false, false);
      }

      // ---- mask; online softmax in log2 domain ---------------------------
      float p0[8], p1[8];
#pragma unroll
      for (int r = 0; r < 8; ++r) {
        const int q   = q0 + r + moff;
        const int kv0 = j + n;
        const int kv1 = j + 16 + n;
        const bool v0 = (q >= kv0) && (q - kv0 <= WIN);
        const bool v1 = (q >= kv1) && (q - kv1 <= WIN);
        p0[r] = v0 ? s0[r] : NEGI;
        p1[r] = v1 ? s1[r] : NEGI;
      }
#pragma unroll
      for (int r = 0; r < 8; ++r) {
        const float mx   = rmax16(fmaxf(p0[r], p1[r]));
        const float mn   = fmaxf(m_i[r], mx);
        const float corr = __builtin_amdgcn_exp2f(m_i[r] - mn);
        p0[r] = __builtin_amdgcn_exp2f(p0[r] - mn);   // masked -> exp2(-1e30)=0
        p1[r] = __builtin_amdgcn_exp2f(p1[r] - mn);
        m_i[r] = mn;
        l_acc[r] *= corr;
#pragma unroll
        for (int dt = 0; dt < 4; ++dt) acc[dt][r] *= corr;
      }

      // ---- P (C/D layout) -> per-wave LDS -> A layout --------------------
      half_t* pl = &lds_p[wave][0];
#pragma unroll
      for (int r = 0; r < 8; ++r) {
        const int m = r + moff;
        pl[m * PSTRIDE +      n] = (half_t)p0[r];
        pl[m * PSTRIDE + 16 + n] = (half_t)p1[r];
      }
      asm volatile("s_wait_dscnt 0" ::: "memory");  // DS in-order per wave
      v16h pa;
      {
        const half_t* row = pl + n * PSTRIDE;
        v8h lo = *reinterpret_cast<const v8h*>(row +      hi * 8);
        v8h hj = *reinterpret_cast<const v8h*>(row + 16 + hi * 8);
#pragma unroll
        for (int i = 0; i < 8; ++i) { pa[i] = lo[i]; pa[8 + i] = hj[i]; }
      }

      // ---- l += P @ ones (row sums, replicated in C/D layout) ------------
      l_acc = __builtin_amdgcn_wmma_f32_16x16x32_f16(false, pa, false, ones, (short)0, l_acc, false, false);

      // ---- O += P @ V : 4 d-tiles from transposed-V LDS ------------------
#pragma unroll
      for (int dt = 0; dt < 4; ++dt) {
        const half_t* vp = &lds_v[cur][(size_t)(16 * dt + n) * VROW_H + 8 * hi];
        acc[dt] = __builtin_amdgcn_wmma_f32_16x16x32_f16(false, pa, false, ld32h(vp), (short)0, acc[dt], false, false);
      }
    }

    __syncthreads();   // all waves done with buf[cur] before it is refilled
  }

  // ---- finalize ----------------------------------------------------------
#pragma unroll
  for (int r = 0; r < 8; ++r) {
    const float inv = 1.0f / l_acc[r];
    const int   m   = r + moff;
    float* orow = Oh + (size_t)(q0 + m) * DH + n;
#pragma unroll
    for (int dt = 0; dt < 4; ++dt) orow[dt * 16] = acc[dt][r] * inv;
  }
}

extern "C" void kernel_launch(void* const* d_in, const int* in_sizes, int n_in,
                              void* d_out, int out_size, void* d_ws, size_t ws_size,
                              hipStream_t stream) {
  (void)in_sizes; (void)n_in; (void)out_size; (void)ws_size;
  const float* Q = (const float*)d_in[0];
  const float* K = (const float*)d_in[1];
  const float* V = (const float*)d_in[2];
  float* O = (float*)d_out;

  half_t* K16  = (half_t*)d_ws;                            // 16 MB
  half_t* Vt16 = K16 + (size_t)NH * T_SEQ * DH;            // 16 MB (needs 32MB ws)

  dim3 pgrid(NH * (T_SEQ / 64));
  hipLaunchKernelGGL(swa_prep_kernel, pgrid, dim3(256), 0, stream, K, V, K16, Vt16);

  dim3 agrid(NH * (T_SEQ / QB));                           // 2048 blocks
  hipLaunchKernelGGL(swa_fwd_kernel, agrid, dim3(WAVES * 32), 0, stream, Q, K16, Vt16, O);
}
